// ScaledDotProductAttention_23364622090566
// MI455X (gfx1250) — matrix operand order repaired and checked by reading, not compiled, not run on hardware
//
#include <hip/hip_runtime.h>
#include <hip/hip_bf16.h>

// Scaled dot-product attention for MI455X (gfx1250), returning (output, attn).
// B=32, S=2048, D=64, fp32 I/O. Memory floor ~580MB (attn write dominates);
// bf16 WMMA (f32 accumulate) keeps compute below the 23.3 TB/s memory floor.

#define BATCH 32
#define SEQ   2048
#define DH    64
// log2(e) / sqrt(64): softmax done in base-2 (native v_exp on AMD is exp2).
#define SCALE2 0.18033688011112042f

typedef __attribute__((ext_vector_type(16))) __bf16 bf16x16;
typedef __attribute__((ext_vector_type(8)))  float  f32x8;

// ---------------------------------------------------------------------------
// Pre-pass: K (fp32, [b][k][d]) -> bf16 [b][k][d]
//           V (fp32, [b][k][d]) -> bf16 transposed [b][d][k]
// Makes every WMMA B-operand a contiguous 32B bf16 load in the main kernel.
// ---------------------------------------------------------------------------
__global__ void cvt_kv_kernel(const float* __restrict__ K,
                              const float* __restrict__ V,
                              __bf16* __restrict__ Kb,
                              __bf16* __restrict__ Vt) {
  size_t i = (size_t)blockIdx.x * blockDim.x + threadIdx.x;
  const size_t n = (size_t)BATCH * SEQ * DH;
  if (i < n) {
    Kb[i] = (__bf16)K[i];
    size_t b = i / ((size_t)SEQ * DH);
    size_t r = i % ((size_t)SEQ * DH);
    size_t d = r / SEQ;
    size_t k = r % SEQ;
    Vt[i] = (__bf16)V[(b * SEQ + k) * DH + d];   // Vt[b][d][k]
  }
}

// ---------------------------------------------------------------------------
// Main kernel: 256 threads = 8 waves per block; each wave owns one 16-query
// tile. Pass 1: online (max, sumexp) per-lane partials over all 128 key
// tiles, merged with shfl_xor. Pass 2: recompute scores, write normalized
// probs to attn (coalesced from WMMA D-layout), LDS-transpose prob tile into
// WMMA A-layout (bf16), accumulate O = P*V with 4 chained bf16 WMMAs.
// ---------------------------------------------------------------------------
__launch_bounds__(256, 1)
__global__ void attn_fused_kernel(const float*  __restrict__ Q,
                                  const __bf16* __restrict__ Kb,
                                  const __bf16* __restrict__ Vt,
                                  float* __restrict__ Out,
                                  float* __restrict__ Attn) {
  __shared__ float pbuf[8][16][33];          // per-wave 16x32 prob tile (+pad)

  const int lane = threadIdx.x & 31;
  const int wave = threadIdx.x >> 5;
  const int g    = blockIdx.x * 8 + wave;    // global 16-query tile id
  const int b    = g >> 7;                   // / (SEQ/16)
  const int qt   = g & 127;
  const int m16  = lane & 15;
  const int half = lane >> 4;
  const int kb8  = half * 8;                 // A-layout K sub-base

  // --- Q tile into WMMA A-layout (bf16), 2 chunks of K=32 features ---------
  const float* Qrow = Q + ((size_t)b * SEQ + qt * 16 + m16) * DH;
  bf16x16 aq[2];
#pragma unroll
  for (int c = 0; c < 2; ++c) {
#pragma unroll
    for (int j = 0; j < 8; ++j) {
      aq[c][j]     = (__bf16)Qrow[c * 32 + half * 16 + j];
      aq[c][j + 8] = (__bf16)Qrow[c * 32 + half * 16 + 8 + j];
    }
  }

  const __bf16* Kbase = Kb + (size_t)b * SEQ * DH;
  const __bf16* Vbase = Vt + (size_t)b * DH * SEQ;

  // --- Pass 1: online softmax stats (per-lane partials over key columns) ---
  float mrow[8], lrow[8];
#pragma unroll
  for (int r = 0; r < 8; ++r) { mrow[r] = -__builtin_inff(); lrow[r] = 0.0f; }

  for (int kt = 0; kt < SEQ / 16; ++kt) {
    const __bf16* kp = Kbase + (size_t)(kt * 16 + m16) * DH + half * 16;
    bf16x16 b0 = *(const bf16x16*)(kp);        // feats  0..31 (lane slice)
    bf16x16 b1 = *(const bf16x16*)(kp + 32);   // feats 32..63 (lane slice)
    f32x8 d = {0.f, 0.f, 0.f, 0.f, 0.f, 0.f, 0.f, 0.f};
    d = __builtin_amdgcn_wmma_f32_16x16x32_bf16(false, aq[0], false, b0,
                                                (short)0, d, false, false);
    d = __builtin_amdgcn_wmma_f32_16x16x32_bf16(false, aq[1], false, b1,
                                                (short)0, d, false, false);
#pragma unroll
    for (int r = 0; r < 8; ++r) {
      float t  = d[r] * SCALE2;
      float mn = fmaxf(mrow[r], t);
      lrow[r]  = lrow[r] * __builtin_amdgcn_exp2f(mrow[r] - mn)
               + __builtin_amdgcn_exp2f(t - mn);
      mrow[r] = mn;
    }
  }

  // Merge 16 per-lane partials per row half (xor 1,2,4,8 stays within half).
#pragma unroll
  for (int r = 0; r < 8; ++r) {
    float m = mrow[r], l = lrow[r];
#pragma unroll
    for (int off = 1; off < 16; off <<= 1) {
      float mo = __shfl_xor(m, off, 32);
      float lo = __shfl_xor(l, off, 32);
      float mn = fmaxf(m, mo);
      l = l * __builtin_amdgcn_exp2f(m - mn)
        + lo * __builtin_amdgcn_exp2f(mo - mn);
      m = mn;
    }
    mrow[r] = m;            // final row max (base-2 domain)
    lrow[r] = 1.0f / l;     // final 1/sumexp
  }

  // --- Pass 2: recompute scores, emit normalized probs, accumulate P*V ----
  f32x8 acc[4];
#pragma unroll
  for (int f = 0; f < 4; ++f) acc[f] = (f32x8){0.f,0.f,0.f,0.f,0.f,0.f,0.f,0.f};

  float* Arow = Attn + ((size_t)b * SEQ + (size_t)qt * 16) * SEQ;

  for (int cc = 0; cc < SEQ / 32; ++cc) {      // 32-key chunks
#pragma unroll
    for (int tt = 0; tt < 2; ++tt) {           // two 16-key score tiles
      const int kt = cc * 2 + tt;
      const __bf16* kp = Kbase + (size_t)(kt * 16 + m16) * DH + half * 16;
      bf16x16 b0 = *(const bf16x16*)(kp);
      bf16x16 b1 = *(const bf16x16*)(kp + 32);
      f32x8 d = {0.f, 0.f, 0.f, 0.f, 0.f, 0.f, 0.f, 0.f};
      d = __builtin_amdgcn_wmma_f32_16x16x32_bf16(false, aq[0], false, b0,
                                                  (short)0, d, false, false);
      d = __builtin_amdgcn_wmma_f32_16x16x32_bf16(false, aq[1], false, b1,
                                                  (short)0, d, false, false);
#pragma unroll
      for (int r = 0; r < 8; ++r) {
        float p = __builtin_amdgcn_exp2f(d[r] * SCALE2 - mrow[r]) * lrow[r];
        const int row = r + half * 8;
        Arow[(size_t)row * SEQ + kt * 16 + m16] = p;   // coalesced 64B/half
        pbuf[wave][row][tt * 16 + m16] = p;            // stage for transpose
      }
    }

    // Intra-wave LDS RAW hazard: wait for this wave's ds stores to land.
    asm volatile("s_wait_dscnt 0" ::: "memory");

    // Read prob tile back in WMMA A-layout, convert to bf16.
    bf16x16 ap;
#pragma unroll
    for (int j = 0; j < 8; ++j) {
      ap[j]     = (__bf16)pbuf[wave][m16][half * 16 + j];
      ap[j + 8] = (__bf16)pbuf[wave][m16][half * 16 + 8 + j];
    }

    // O[16x64] += P[16x32] * V[32x64]  (4 N-tiles of 16 features)
#pragma unroll
    for (int f = 0; f < 4; ++f) {
      const __bf16* vp = Vbase + (size_t)(f * 16 + m16) * SEQ
                       + cc * 32 + half * 16;
      bf16x16 bv = *(const bf16x16*)vp;        // 16 consecutive keys (32B)
      acc[f] = __builtin_amdgcn_wmma_f32_16x16x32_bf16(false, ap, false, bv,
                                                       (short)0, acc[f],
                                                       false, false);
    }
  }

  // --- Store output tile (f32, coalesced 64B/half-wave) --------------------
  float* Orow = Out + ((size_t)b * SEQ + (size_t)qt * 16) * DH;
#pragma unroll
  for (int f = 0; f < 4; ++f) {
#pragma unroll
    for (int r = 0; r < 8; ++r) {
      Orow[(size_t)(r + half * 8) * DH + f * 16 + m16] = acc[f][r];
    }
  }
}

// ---------------------------------------------------------------------------
extern "C" void kernel_launch(void* const* d_in, const int* in_sizes, int n_in,
                              void* d_out, int out_size, void* d_ws,
                              size_t ws_size, hipStream_t stream) {
  const float* q = (const float*)d_in[0];
  const float* k = (const float*)d_in[1];
  const float* v = (const float*)d_in[2];

  float* out  = (float*)d_out;                         // [B,S,D] first
  float* attn = out + (size_t)BATCH * SEQ * DH;        // then [B,S,S]

  // Workspace: 8MB bf16 K + 8MB bf16 V^T (requires ws_size >= 16 MiB).
  __bf16* kb = (__bf16*)d_ws;
  __bf16* vt = kb + (size_t)BATCH * SEQ * DH;

  const size_t n = (size_t)BATCH * SEQ * DH;
  cvt_kv_kernel<<<(unsigned)((n + 255) / 256), 256, 0, stream>>>(k, v, kb, vt);

  // 4096 query tiles of 16, 8 waves (tiles) per 256-thread block.
  attn_fused_kernel<<<(BATCH * (SEQ / 16)) / 8, 256, 0, stream>>>(
      q, kb, vt, out, attn);
}
